// SDT_45741401702572
// MI455X (gfx1250) — matrix operand order repaired and checked by reading, not compiled, not run on hardware
//
#include <hip/hip_runtime.h>
#include <hip/hip_bf16.h>

typedef __attribute__((ext_vector_type(16))) _Float16 v16h;
typedef __attribute__((ext_vector_type(8)))  _Float16 v8h;
typedef __attribute__((ext_vector_type(8)))  float    v8f;

#define B_ROWS  16384
#define D_IN    128
#define HID     16
#define N_INNER 255
#define N_LEAF  256

// ---------------------------------------------------------------------------
// Kernel 1a: split x (fp32) into f16 hi + lo residual (captures ~22 mantissa
// bits; the dropped lo*lo WMMA term is ~2^-22 relative -> fp32-grade accuracy)
// ---------------------------------------------------------------------------
__global__ void split_x_kernel(const float* __restrict__ x,
                               _Float16* __restrict__ xhi,
                               _Float16* __restrict__ xlo, int n) {
  int i = blockIdx.x * blockDim.x + threadIdx.x;
  if (i < n) {
    float v = x[i];
    _Float16 h = (_Float16)v;
    xhi[i] = h;
    xlo[i] = (_Float16)(v - (float)h);
  }
}

// ---------------------------------------------------------------------------
// Kernel 1b: W1 (n, k, h) fp32 -> w_pack: per node one contiguous 8KB block
//   [ hi tile (h,k) 16x128 | lo tile (h,k) 16x128 ]  (transposed to (h,k) so
// each WMMA B-fragment lane reads two contiguous 16-byte runs)
// ---------------------------------------------------------------------------
__global__ void split_w_kernel(const float* __restrict__ W1,
                               _Float16* __restrict__ w_pack, int n) {
  int i = blockIdx.x * blockDim.x + threadIdx.x;   // i = node*2048 + h*128 + k
  if (i < n) {
    int k    = i & 127;
    int h    = (i >> 7) & 15;
    int node = i >> 11;
    float v = W1[(size_t)node * (D_IN * HID) + (size_t)k * HID + h];
    _Float16 hh = (_Float16)v;
    _Float16 ll = (_Float16)(v - (float)hh);
    _Float16* dst = w_pack + (size_t)node * 4096 + (size_t)h * 128 + k;
    dst[0]    = hh;
    dst[2048] = ll;
  }
}

// ---------------------------------------------------------------------------
// Async staging helpers (CDNA5 GLOBAL_LOAD_ASYNC_TO_LDS, tracked by ASYNCcnt)
// ---------------------------------------------------------------------------
__device__ __forceinline__ unsigned lds_off(const void* p) {
  // LDS generic addresses map to DS space by discarding the upper bits
  return (unsigned)(uintptr_t)p;
}

// Copy one 8192-byte node tile global -> LDS: 16 async b128 ops x 32 lanes
__device__ __forceinline__ void stage_async(const _Float16* __restrict__ gsrc,
                                            unsigned lbase, int lane) {
  const char* g = (const char*)gsrc + lane * 16;
  unsigned    l = lbase + lane * 16;
#pragma unroll
  for (int it = 0; it < 16; ++it) {
    asm volatile("global_load_async_to_lds_b128 %0, %1, off"
                 :: "v"(l + it * 512), "v"(g + (size_t)it * 512)
                 : "memory");
  }
}

// ---------------------------------------------------------------------------
// Kernel 2: per wave, one 16-row x 16-col (= one node's HID) C tile.
//   H = relu(X @ W1[n] + b1[n]);  p[:,n] = sigmoid(H @ W2[n] + b2[n])
// A fragments (x rows) live in VGPRs across the node loop; the per-node B
// tile (hi+lo) is double-buffered through LDS by wave 0 with async copies
// overlapping the other waves' WMMA work.
// ---------------------------------------------------------------------------
__global__ void __launch_bounds__(256)
gemm_sigmoid_kernel(const _Float16* __restrict__ xhi,
                    const _Float16* __restrict__ xlo,
                    const _Float16* __restrict__ w_pack,
                    const float* __restrict__ b1,
                    const float* __restrict__ W2,
                    const float* __restrict__ b2,
                    float* __restrict__ p_out) {
  __shared__ _Float16 smem[2][4096];     // 2 x 8KB node tiles (hi|lo)

  const int lane    = threadIdx.x & 31;
  const int wave    = threadIdx.x >> 5;
  const int rowBase = blockIdx.x * 128 + wave * 16;
  const int m       = rowBase + (lane & 15);
  // ISA 16-bit A layout: lane half (bit4) selects K group +8 within each 16
  const int kb_a = (lane & 16) ? 8 : 0;
  // ISA 16-bit B layout: lane half (bit4) selects K group +8 within each 16
  const int kb_b = (lane & 16) ? 8 : 0;
  const int hcol = lane & 15;            // N dimension of C = HID column

  // Preload A fragments: 4 chunks of K=32, hi and lo (stay resident)
  v16h ahi[4], alo[4];
#pragma unroll
  for (int c = 0; c < 4; ++c) {
    const _Float16* pah = xhi + (size_t)m * D_IN + c * 32 + kb_a;
    const _Float16* pal = xlo + (size_t)m * D_IN + c * 32 + kb_a;
    v8h h0 = *(const v8h*)(pah);         // K = kb_a + 0..7
    v8h h1 = *(const v8h*)(pah + 16);    // K = 16 + kb_a + 0..7
    v8h l0 = *(const v8h*)(pal);
    v8h l1 = *(const v8h*)(pal + 16);
#pragma unroll
    for (int j = 0; j < 8; ++j) {
      ahi[c][j] = h0[j]; ahi[c][8 + j] = h1[j];
      alo[c][j] = l0[j]; alo[c][8 + j] = l1[j];
    }
  }

  const int nBegin = blockIdx.y * 51;    // 5 * 51 == 255 nodes
  if (wave == 0)
    stage_async(w_pack + (size_t)nBegin * 4096, lds_off(&smem[0][0]), lane);

  for (int i = 0; i < 51; ++i) {
    const int n = nBegin + i;
    if (wave == 0)
      asm volatile("s_wait_asynccnt 0x0" ::: "memory");
    __syncthreads();                     // staged tile visible to all waves;
                                         // prior reads of the other buffer done
    if (wave == 0 && i + 1 < 51)
      stage_async(w_pack + (size_t)(n + 1) * 4096,
                  lds_off(&smem[(i + 1) & 1][0]), lane);

    const _Float16* wb = &smem[i & 1][hcol * 128 + kb_b];
    v8f acc = {};
#pragma unroll
    for (int c = 0; c < 4; ++c) {
      v8h bh0 = *(const v8h*)(wb + c * 32);
      v8h bh1 = *(const v8h*)(wb + c * 32 + 16);
      v8h bl0 = *(const v8h*)(wb + 2048 + c * 32);
      v8h bl1 = *(const v8h*)(wb + 2048 + c * 32 + 16);
      v16h bh, bl;
#pragma unroll
      for (int j = 0; j < 8; ++j) {
        bh[j] = bh0[j]; bh[8 + j] = bh1[j];
        bl[j] = bl0[j]; bl[8 + j] = bl1[j];
      }
      acc = __builtin_amdgcn_wmma_f32_16x16x32_f16(false, ahi[c], false, bh,
                                                   (short)0, acc, false, false);
      acc = __builtin_amdgcn_wmma_f32_16x16x32_f16(false, ahi[c], false, bl,
                                                   (short)0, acc, false, false);
      acc = __builtin_amdgcn_wmma_f32_16x16x32_f16(false, alo[c], false, bh,
                                                   (short)0, acc, false, false);
    }

    // Epilogue: relu(+b1), dot with W2 across the lane (N) dimension, sigmoid
    float b1v = b1[n * HID + hcol];
    float w2v = W2[n * HID + hcol];
    float b2v = b2[n];
    float s[8];
#pragma unroll
    for (int r = 0; r < 8; ++r) {
      float t = acc[r] + b1v;
      t = t > 0.f ? t : 0.f;
      t *= w2v;
      t += __shfl_xor(t, 1, 16);
      t += __shfl_xor(t, 2, 16);
      t += __shfl_xor(t, 4, 16);
      t += __shfl_xor(t, 8, 16);
      s[r] = t;
    }
    if (hcol == 0) {                     // lanes 0 and 16 hold row sums
      int roff = (lane & 16) ? 8 : 0;    // C layout: upper lane half -> M+8
#pragma unroll
      for (int r = 0; r < 8; ++r) {
        float z  = s[r] + b2v;
        float pv = 1.f / (1.f + __expf(-z));
        p_out[(size_t)(rowBase + roff + r) * N_INNER + n] = pv;
      }
    }
  }
}

// ---------------------------------------------------------------------------
// Kernel 3: tree recursion. One wave per batch row; lanes stride over nodes
// and leaves computing heap-ancestor products (child even -> 1-p, odd -> p).
// ---------------------------------------------------------------------------
__global__ void __launch_bounds__(256)
tree_kernel(const float* __restrict__ p, const float* __restrict__ leaf,
            float* __restrict__ h_out, float* __restrict__ path_probs,
            float* __restrict__ node_reach) {
  const int gwave = (int)((blockIdx.x * blockDim.x + threadIdx.x) >> 5);
  const int lane  = threadIdx.x & 31;
  const float* prow = p + (size_t)gwave * N_INNER;

  // node_reach: product of edge probs from root down to (not including) node
  for (int mm = lane; mm < N_INNER; mm += 32) {
    int v = mm + 1;                      // heap index (root = 1)
    float pr = 1.f;
    for (int c = v; c > 1; c >>= 1) {
      float pa = prow[(c >> 1) - 1];
      pr *= (c & 1) ? pa : (1.f - pa);
    }
    node_reach[(size_t)gwave * N_INNER + mm] = pr;
  }

  // path_probs over the 256 leaves + h = path_probs . leaf
  float hacc = 0.f;
  for (int j = lane; j < N_LEAF; j += 32) {
    int v = N_LEAF + j;                  // leaf heap index
    float pr = 1.f;
#pragma unroll
    for (int k = 0; k < 8; ++k) {
      float pa = prow[(v >> 1) - 1];
      pr *= (v & 1) ? pa : (1.f - pa);
      v >>= 1;
    }
    path_probs[(size_t)gwave * N_LEAF + j] = pr;
    hacc += pr * leaf[j];
  }
#pragma unroll
  for (int off = 16; off > 0; off >>= 1) hacc += __shfl_xor(hacc, off, 32);
  if (lane == 0) h_out[gwave] = hacc;
}

// ---------------------------------------------------------------------------
extern "C" void kernel_launch(void* const* d_in, const int* in_sizes, int n_in,
                              void* d_out, int out_size, void* d_ws, size_t ws_size,
                              hipStream_t stream) {
  const float* x    = (const float*)d_in[0];
  const float* W1   = (const float*)d_in[1];
  const float* b1   = (const float*)d_in[2];
  const float* W2   = (const float*)d_in[3];
  const float* b2   = (const float*)d_in[4];
  const float* leaf = (const float*)d_in[5];

  // Output tuple layout (concatenated flat, fp32):
  //   h (B), path_probs (B,256), p (B,255), node_reach (B,255)
  float* out        = (float*)d_out;
  float* h_out      = out;
  float* path_probs = out + B_ROWS;
  float* p_out      = path_probs + (size_t)B_ROWS * N_LEAF;
  float* node_reach = p_out + (size_t)B_ROWS * N_INNER;

  // Workspace: f16 hi/lo copies of x, packed per-node W tiles (~10.5 MB)
  _Float16* xhi    = (_Float16*)d_ws;
  _Float16* xlo    = xhi + (size_t)B_ROWS * D_IN;
  _Float16* w_pack = xlo + (size_t)B_ROWS * D_IN;   // 255 * 4096 halves

  int nx = B_ROWS * D_IN;
  split_x_kernel<<<(nx + 255) / 256, 256, 0, stream>>>(x, xhi, xlo, nx);
  int nw = N_INNER * D_IN * HID;
  split_w_kernel<<<(nw + 255) / 256, 256, 0, stream>>>(W1, w_pack, nw);

  dim3 g2(B_ROWS / 128, 5);   // 8 waves x 16 rows per WG; 5 node groups of 51
  gemm_sigmoid_kernel<<<g2, 256, 0, stream>>>(xhi, xlo, w_pack,
                                              b1, W2, b2, p_out);

  tree_kernel<<<(B_ROWS * 32) / 256, 256, 0, stream>>>(p_out, leaf, h_out,
                                                       path_probs, node_reach);
}
